// TRPN_73349451481376
// MI455X (gfx1250) — matrix operand order repaired and checked by reading, not compiled, not run on hardware
//
#include <hip/hip_runtime.h>
#include <hip/hip_bf16.h>

#define T_  4
#define N_  400
#define S_  395
#define D_  128
#define H0_ 256
#define H1_ 128
#define GD_ 768

typedef __attribute__((ext_vector_type(16))) _Float16 v16h;
typedef __attribute__((ext_vector_type(8)))  _Float16 v8h;
typedef __attribute__((ext_vector_type(8)))  float    v8f;

// ---------------------------------------------------------------------------
// Stage 0a: per-node first-layer projections.
//   Af[t,i,h] = node_feat[t,i,:] . fc1_w1[h, 0:128]   + fc1_b1[h]
//   Bf[t,j,h] = node_feat[t,j,:] . fc1_w1[h, 128:256]
// so layer-1 activation for pair (a,b) is relu(Af[a,:] + Bf[b,:]).
// ---------------------------------------------------------------------------
__global__ __launch_bounds__(256) void k_node_proj(
    const float* __restrict__ nf, const float* __restrict__ w1,
    const float* __restrict__ b1, _Float16* __restrict__ Af,
    _Float16* __restrict__ Bf)
{
  int idx = blockIdx.x * 256 + threadIdx.x;
  if (idx >= T_ * N_ * H0_) return;
  int h  = idx & (H0_ - 1);
  int tn = idx >> 8;
  const float* x  = nf + (size_t)tn * D_;
  const float* wa = w1 + (size_t)h * (2 * D_);
  float sa = 0.f, sb = 0.f;
#pragma unroll 4
  for (int k = 0; k < D_; ++k) {
    float xv = x[k];
    sa += xv * wa[k];
    sb += xv * wa[D_ + k];
  }
  Af[idx] = (_Float16)(sa + b1[h]);
  Bf[idx] = (_Float16)sb;
}

// ---------------------------------------------------------------------------
// Stage 0b: pre-swizzle fc1_w2 (H1=128 x H0=256) into WMMA B-operand register
// image (32x16 f16 per (ks,nt) tile): col n=lane&15, K=8*(lane>>4)+e for e<8,
// K=16+8*(lane>>4)+(e-8) for e>=8  (ISA 7.12.2 / 7.12.4 B layout).
// frag[((ks*8+nt)*32 + lane)*16 + e] = w2[n, k]
// ---------------------------------------------------------------------------
__global__ __launch_bounds__(256) void k_w2frag(const float* __restrict__ w2,
                                                _Float16* __restrict__ frag)
{
  int idx = blockIdx.x * 256 + threadIdx.x;
  if (idx >= 8 * 8 * 32 * 16) return;
  int e     = idx & 15;
  int lane  = (idx >> 4) & 31;
  int tilei = idx >> 9;               // ks*8 + nt
  int ks = tilei >> 3, nt = tilei & 7;
  int n = nt * 16 + (lane & 15);
  int k = ks * 32 + (e >> 3) * 16 + (lane >> 4) * 8 + (e & 7);
  frag[idx] = (_Float16)w2[(size_t)n * H0_ + k];
}

// ---------------------------------------------------------------------------
// Stage 1 (dominant GEMM): fused pair MLP via WMMA.
// Each wave handles one (t, a, b-block of 16) tile: M=16 pairs, K=256, N=128.
// Per K-step: pack A fragment with v_pk_add_f16 + v_pk_max (ReLU), bulk-load
// all 8 B fragments from LDS, then 8 back-to-back v_wmma_f32_16x16x32_f16.
// Epilogue: +b2, relu, dot with w3 (cross-lane reduce), sigmoid, write
// learned_score and masked learned_adj.
// ---------------------------------------------------------------------------
__global__ __launch_bounds__(256) void k_pair_mlp(
    const _Float16* __restrict__ Af, const _Float16* __restrict__ Bf,
    const _Float16* __restrict__ Wfrag,
    const float* __restrict__ b2v, const float* __restrict__ w3v,
    const float* __restrict__ b3v, const float* __restrict__ adj,
    float* __restrict__ score_out, float* __restrict__ adjL)
{
  __shared__ __align__(32) _Float16 ldsW[8 * 8 * 32 * 16];   // 64 KB
  int tid = threadIdx.x;
  {
    const uint4* src = (const uint4*)Wfrag;   // 4096 uint4 total
    uint4* dst = (uint4*)ldsW;
#pragma unroll
    for (int i = 0; i < 16; ++i) dst[tid + 256 * i] = src[tid + 256 * i];
  }
  __syncthreads();

  int wave = tid >> 5;
  int lane = tid & 31;
  int tile = blockIdx.x * 8 + wave;           // 5000 * 8 == 40000 tiles
  if (tile >= T_ * N_ * (N_ / 16)) return;

  int bblk = tile % (N_ / 16);
  int ta   = tile / (N_ / 16);
  int a    = ta % N_;
  int t    = ta / N_;
  int b0   = bblk * 16;
  int m    = lane & 15;
  int hi   = lane >> 4;

  const _Float16* Arow = Af + ((size_t)(t * N_ + a)) * H0_;
  const _Float16* Brow = Bf + ((size_t)(t * N_ + b0 + m)) * H0_;

  v8f acc[8];
#pragma unroll
  for (int ntt = 0; ntt < 8; ++ntt)
#pragma unroll
    for (int e = 0; e < 8; ++e) acc[ntt][e] = 0.f;

  v8h zv;
#pragma unroll
  for (int e = 0; e < 8; ++e) zv[e] = (_Float16)0.0f;

  for (int ks = 0; ks < 8; ++ks) {
    int off1 = ks * 32 + hi * 8;
    __builtin_prefetch(Brow + off1 + 32, 0, 3);
    v8h a1  = *(const v8h*)(Arow + off1);
    v8h bb1 = *(const v8h*)(Brow + off1);
    v8h a2  = *(const v8h*)(Arow + off1 + 16);
    v8h bb2 = *(const v8h*)(Brow + off1 + 16);
    v8h h1 = __builtin_elementwise_max(a1 + bb1, zv);   // v_pk_add + v_pk_max
    v8h h2 = __builtin_elementwise_max(a2 + bb2, zv);
    v16h afr = __builtin_shufflevector(h1, h2, 0, 1, 2, 3, 4, 5, 6, 7,
                                       8, 9, 10, 11, 12, 13, 14, 15);

    // bulk-load all B fragments for this K-step, then issue WMMAs back-to-back
    v16h bfr[8];
#pragma unroll
    for (int ntt = 0; ntt < 8; ++ntt)
      bfr[ntt] = *(const v16h*)(ldsW + (((ks * 8 + ntt) * 32 + lane) * 16));
#pragma unroll
    for (int ntt = 0; ntt < 8; ++ntt)
      acc[ntt] = __builtin_amdgcn_wmma_f32_16x16x32_f16(
          false, afr, false, bfr[ntt], (short)0, acc[ntt], false, false);
  }

  // Epilogue: lane holds column n = ntt*16 + (lane&15); VGPR r -> row m = r + 8*hi.
  float p[8];
#pragma unroll
  for (int r = 0; r < 8; ++r) p[r] = 0.f;
#pragma unroll
  for (int ntt = 0; ntt < 8; ++ntt) {
    int n = ntt * 16 + (lane & 15);
    float bb = b2v[n];
    float w  = w3v[n];
#pragma unroll
    for (int r = 0; r < 8; ++r) {
      float g = acc[ntt][r] + bb;
      g = g > 0.f ? g : 0.f;
      p[r] += g * w;
    }
  }
  // reduce across the 16 lanes of each half-wave (rows 0-7 in lanes 0-15,
  // rows 8-15 in lanes 16-31)
#pragma unroll
  for (int mask = 1; mask < 16; mask <<= 1)
#pragma unroll
    for (int r = 0; r < 8; ++r) p[r] += __shfl_xor(p[r], mask, 32);

  if ((lane & 15) == 0) {
    float b3s = b3v[0];
#pragma unroll
    for (int r = 0; r < 8; ++r) {
      int b = b0 + hi * 8 + r;
      float s = 1.f / (1.f + __expf(-(p[r] + b3s)));
      size_t idx = ((size_t)t * N_ + a) * N_ + b;
      score_out[idx] = s;
      float la = s;
      if (a < S_ && b < S_) {
        float av = adj[((size_t)t * S_ + a) * S_ + b];
        la = (av > 0.f) ? 1.0f : -s;
      }
      adjL[idx] = la;
    }
  }
}

// ---------------------------------------------------------------------------
// Stage 2a: constant part of support (query-flat columns, independent of n).
// C[t,o] = sum_{j<640} node_feat[t, S + j/128, j%128] * weight_G[128+j, o]
// ---------------------------------------------------------------------------
__global__ __launch_bounds__(256) void k_gcn_const(
    const float* __restrict__ nf, const float* __restrict__ wG,
    float* __restrict__ C)
{
  int idx = blockIdx.x * 256 + threadIdx.x;
  if (idx >= T_ * GD_) return;
  int o = idx % GD_;
  int t = idx / GD_;
  const float* q = nf + ((size_t)t * N_ + S_) * D_;  // 640 contiguous floats
  float s = 0.f;
  for (int j = 0; j < (N_ - S_) * D_; ++j)
    s += q[j] * wG[(size_t)(D_ + j) * GD_ + o];
  C[idx] = s;
}

// ---------------------------------------------------------------------------
// Stage 2b: support[t,n,o] = node_feat[t,n,:] . weight_G[0:128, o] + C[t,o]
// ---------------------------------------------------------------------------
__global__ __launch_bounds__(256) void k_support(
    const float* __restrict__ nf, const float* __restrict__ wG,
    const float* __restrict__ C, float* __restrict__ sup)
{
  int idx = blockIdx.x * 256 + threadIdx.x;
  if (idx >= T_ * N_ * GD_) return;
  int o  = idx % GD_;
  int tn = idx / GD_;
  int t  = tn / N_;
  const float* x = nf + (size_t)tn * D_;
  float s = C[t * GD_ + o];
#pragma unroll 4
  for (int f = 0; f < D_; ++f) s += x[f] * wG[(size_t)f * GD_ + o];
  sup[idx] = s;
}

// ---------------------------------------------------------------------------
// Stage 3: wl[t,n,o] = relu( learned_adj[t,n,:] . support[t,:,o] + bias_G[o] )
// ---------------------------------------------------------------------------
__global__ __launch_bounds__(256) void k_wl(
    const float* __restrict__ adjL, const float* __restrict__ sup,
    const float* __restrict__ bG, float* __restrict__ wl)
{
  int idx = blockIdx.x * 256 + threadIdx.x;
  if (idx >= T_ * N_ * GD_) return;
  int o  = idx % GD_;
  int tn = idx / GD_;
  int t  = tn / N_;
  const float* arow = adjL + (size_t)tn * N_;
  const float* scol = sup + ((size_t)t * N_) * GD_ + o;
  float s = bG[o];
#pragma unroll 4
  for (int mm = 0; mm < N_; ++mm) s += arow[mm] * scol[(size_t)mm * GD_];
  wl[idx] = fmaxf(s, 0.f);
}

// ---------------------------------------------------------------------------
// Stage 4a: output 1 = mean over t of wl  ->  (400, 768)
// ---------------------------------------------------------------------------
__global__ __launch_bounds__(256) void k_mean(const float* __restrict__ wl,
                                              float* __restrict__ out)
{
  int idx = blockIdx.x * 256 + threadIdx.x;
  if (idx >= N_ * GD_) return;
  float s = 0.f;
#pragma unroll
  for (int t = 0; t < T_; ++t) s += wl[(size_t)t * N_ * GD_ + idx];
  out[idx] = s * 0.25f;
}

// ---------------------------------------------------------------------------
// Stage 4b: fc2 head per (t,n): 768->128->128->5, sigmoid, write (t,c,n).
// One block of 128 threads per (t,n) row, LDS-staged activations.
// ---------------------------------------------------------------------------
__global__ __launch_bounds__(128) void k_fc2(
    const float* __restrict__ wl, const float* __restrict__ w1,
    const float* __restrict__ b1, const float* __restrict__ w2,
    const float* __restrict__ b2, const float* __restrict__ w3,
    const float* __restrict__ b3, float* __restrict__ out_qs)
{
  __shared__ float sh_in[GD_];
  __shared__ float sh_h[H1_];
  __shared__ float sh_g[H1_];
  int tn = blockIdx.x;                 // 0 .. T*N-1
  int t  = tn / N_;
  int nn = tn % N_;
  int h  = threadIdx.x;

  for (int i = h; i < GD_; i += 128) sh_in[i] = wl[(size_t)tn * GD_ + i];
  __syncthreads();

  float s = b1[h];
  for (int o = 0; o < GD_; ++o) s += sh_in[o] * w1[(size_t)h * GD_ + o];
  sh_h[h] = fmaxf(s, 0.f);
  __syncthreads();

  float s2 = b2[h];
#pragma unroll 4
  for (int k = 0; k < H1_; ++k) s2 += sh_h[k] * w2[h * H1_ + k];
  sh_g[h] = fmaxf(s2, 0.f);
  __syncthreads();

  if (h < 5) {
    float s3 = b3[h];
#pragma unroll 4
    for (int k = 0; k < H1_; ++k) s3 += sh_g[k] * w3[h * H1_ + k];
    float sig = 1.f / (1.f + __expf(-s3));
    out_qs[((size_t)t * 5 + h) * N_ + nn] = sig;
  }
}

// ---------------------------------------------------------------------------
extern "C" void kernel_launch(void* const* d_in, const int* in_sizes, int n_in,
                              void* d_out, int out_size, void* d_ws,
                              size_t ws_size, hipStream_t stream) {
  (void)in_sizes; (void)n_in; (void)out_size; (void)ws_size;

  const float* nf     = (const float*)d_in[0];
  const float* adj    = (const float*)d_in[1];
  const float* fc1_w1 = (const float*)d_in[2];
  const float* fc1_b1 = (const float*)d_in[3];
  const float* fc1_w2 = (const float*)d_in[4];
  const float* fc1_b2 = (const float*)d_in[5];
  const float* fc1_w3 = (const float*)d_in[6];
  const float* fc1_b3 = (const float*)d_in[7];
  const float* wG     = (const float*)d_in[8];
  const float* bG     = (const float*)d_in[9];
  const float* fc2_w1 = (const float*)d_in[10];
  const float* fc2_b1 = (const float*)d_in[11];
  const float* fc2_w2 = (const float*)d_in[12];
  const float* fc2_b2 = (const float*)d_in[13];
  const float* fc2_w3 = (const float*)d_in[14];
  const float* fc2_b3 = (const float*)d_in[15];

  // workspace layout (all offsets multiples of 256 B)
  char* ws = (char*)d_ws;
  _Float16* Af    = (_Float16*)(ws + 0);          //   819200 B
  _Float16* Bf    = (_Float16*)(ws + 819200);     //   819200 B
  _Float16* Wfrag = (_Float16*)(ws + 1638400);    //    65536 B
  float*    adjL  = (float*)(ws + 1703936);       //  2560000 B
  float*    Cbuf  = (float*)(ws + 4263936);       //    12288 B
  float*    sup   = (float*)(ws + 4276224);       //  4915200 B
  float*    wl    = (float*)(ws + 9191424);       //  4915200 B  (end ~13.5 MB)

  float* out       = (float*)d_out;
  float* out_mean  = out;                         // 400*768
  float* out_score = out + (size_t)N_ * GD_;      // T*N*N
  float* out_qs    = out_score + (size_t)T_ * N_ * N_;  // T*5*N

  // Stage 0
  k_node_proj<<<(T_ * N_ * H0_ + 255) / 256, 256, 0, stream>>>(
      nf, fc1_w1, fc1_b1, Af, Bf);
  k_w2frag<<<(8 * 8 * 32 * 16 + 255) / 256, 256, 0, stream>>>(fc1_w2, Wfrag);

  // Stage 1 (WMMA): 40000 wave-tiles, 8 waves per block
  k_pair_mlp<<<T_ * N_ * (N_ / 16) / 8, 256, 0, stream>>>(
      Af, Bf, Wfrag, fc1_b2, fc1_w3, fc1_b3, adj, out_score, adjL);

  // Stage 2
  k_gcn_const<<<(T_ * GD_ + 255) / 256, 256, 0, stream>>>(nf, wG, Cbuf);
  k_support<<<(T_ * N_ * GD_ + 255) / 256, 256, 0, stream>>>(nf, wG, Cbuf, sup);

  // Stage 3
  k_wl<<<(T_ * N_ * GD_ + 255) / 256, 256, 0, stream>>>(adjL, sup, bG, wl);

  // Stage 4
  k_mean<<<(N_ * GD_ + 255) / 256, 256, 0, stream>>>(wl, out_mean);
  k_fc2<<<T_ * N_, 128, 0, stream>>>(wl, fc2_w1, fc2_b1, fc2_w2, fc2_b2,
                                     fc2_w3, fc2_b3, out_qs);
}
